// Graph_GAT_10857677325007
// MI455X (gfx1250) — matrix operand order repaired and checked by reading, not compiled, not run on hardware
//
#include <hip/hip_runtime.h>
#include <hip/hip_bf16.h>
#include <math.h>

// ---------------------------------------------------------------------------
// Graph_GAT DTA forward for gfx1250 (MI455X). Dense matmuls -> WMMA f16/f32
// (128x64 block tile, 2x2 WMMA tiles per wave32, vectorized LDS fragments,
// branch-free clamped staging), graph scatter ops -> bandwidth-bound atomics.
// ---------------------------------------------------------------------------

typedef __attribute__((ext_vector_type(16))) _Float16 v16h;
typedef __attribute__((ext_vector_type(8)))  _Float16 v8h;
typedef __attribute__((ext_vector_type(4)))  _Float16 v4h;
typedef __attribute__((ext_vector_type(8)))  float    v8f;

#define NM 30000
#define EM 90000
#define NP 60000
#define EP 300000
#define NB 128

static inline int nblk(long long n) { return (int)((n + 255) / 256); }

// ------------------------------- device utils ------------------------------

__device__ __forceinline__ void atomicMaxF(float* addr, float val) {
  unsigned* ua = (unsigned*)addr;
  unsigned old = *ua;
  while (__uint_as_float(old) < val) {
    unsigned assumed = old;
    old = atomicCAS(ua, assumed, __float_as_uint(val));
    if (old == assumed) break;
  }
}

__global__ void fill_kernel(float* __restrict__ p, float v, long long n) {
  long long i = (long long)blockIdx.x * blockDim.x + threadIdx.x;
  long long st = (long long)gridDim.x * blockDim.x;
  for (; i < n; i += st) p[i] = v;
}

// ------------------------- WMMA tiled GEMM (fp32 io) -----------------------
// C[M,N] = act( A[M,K] @ W[K,N] + bias[N] ), act: 0=none 1=relu 2=tanh
// Block: 256 threads = 8 wave32 (4x2). Block tile 128x64, wave tile 32x32
// (2x2 v_wmma_f32_16x16x32_f16 per k-step). A staged row-major, B staged
// transposed so every lane's fragment is contiguous 16B LDS runs. Staging is
// branch-free: indices are clamped in-range and OOB lanes zeroed via select.

#define GEMM_BM 128
#define GEMM_BN 64
#define GEMM_BK 32
#define LDS_STR 40   // 32 + 8 halves padding; 80B row stride keeps 16B alignment

// two contiguous 8-half runs -> one 16-half fragment
__device__ __forceinline__ v16h frag_cat(const _Float16* p0, const _Float16* p1) {
  v8h x = *(const v8h*)p0;
  v8h y = *(const v8h*)p1;
  return __builtin_shufflevector(x, y, 0, 1, 2, 3, 4, 5, 6, 7,
                                 8, 9, 10, 11, 12, 13, 14, 15);
}

__global__ __launch_bounds__(256) void gemm_wmma_kernel(
    const float* __restrict__ A, const float* __restrict__ Wt,
    const float* __restrict__ bias, float* __restrict__ C_,
    int M, int N, int K, int act)
{
  __shared__ __align__(16) _Float16 As[GEMM_BM][LDS_STR];   // 10240 B
  __shared__ __align__(16) _Float16 Bst[GEMM_BN][LDS_STR];  //  5120 B (transposed: [n][k])

  const int tid  = threadIdx.x;
  const int lane = tid & 31;
  const int l15  = lane & 15;
  const int wid  = tid >> 5;      // 0..7
  const int wm   = wid >> 1;      // 0..3  -> 32 rows each
  const int wn   = wid & 1;       // 0..1  -> 32 cols each
  const int m0   = blockIdx.y * GEMM_BM;
  const int n0   = blockIdx.x * GEMM_BN;

  const bool a_vec = ((K & 3) == 0);   // float4 path legal & aligned
  const bool b_vec = ((N & 3) == 0);

  v8f acc[2][2] = {{{}, {}}, {{}, {}}};
  const int ksteps = (K + GEMM_BK - 1) / GEMM_BK;

  for (int ks = 0; ks < ksteps; ++ks) {
    const int k0 = ks * GEMM_BK;

    // ---- stage A tile: 128x32 floats = 1024 float4 chunks, 4 per thread ----
    // Branch-free: clamp indices, load unconditionally, zero OOB via select.
    #pragma unroll
    for (int i = 0; i < 4; ++i) {
      int idx = tid + i * 256;          // 0..1023
      int r = idx >> 3;                 // 0..127
      int c = (idx & 7) * 4;            // 0..28
      int gm = m0 + r, gk = k0 + c;
      bool mok = gm < M;
      int gmc = mok ? gm : (M - 1);
      v4h hv;
      if (a_vec && gk + 4 <= K) {       // uniform for all our layer shapes
        const float4 f = *(const float4*)(A + (long long)gmc * K + gk);
        hv[0] = mok ? (_Float16)f.x : (_Float16)0.0f;
        hv[1] = mok ? (_Float16)f.y : (_Float16)0.0f;
        hv[2] = mok ? (_Float16)f.z : (_Float16)0.0f;
        hv[3] = mok ? (_Float16)f.w : (_Float16)0.0f;
      } else {
        #pragma unroll
        for (int j = 0; j < 4; ++j) {
          bool kok = (gk + j) < K;
          int gkc = kok ? (gk + j) : (K - 1);
          float v = A[(long long)gmc * K + gkc];
          hv[j] = (mok && kok) ? (_Float16)v : (_Float16)0.0f;
        }
      }
      *(v4h*)&As[r][c] = hv;
    }

    // ---- stage B tile transposed: W[K,N] 32x64 -> Bst[n][k], 2 float4/thread
    #pragma unroll
    for (int i = 0; i < 2; ++i) {
      int idx = tid + i * 256;          // 0..511
      int r = idx >> 4;                 // k row 0..31
      int c = (idx & 15) * 4;           // n col 0..60
      int gk = k0 + r, gn = n0 + c;
      bool kok = gk < K;
      int gkc = kok ? gk : (K - 1);
      v4h hv;
      if (b_vec && gn + 4 <= N) {       // uniform for all our layer shapes
        const float4 f = *(const float4*)(Wt + (long long)gkc * N + gn);
        hv[0] = kok ? (_Float16)f.x : (_Float16)0.0f;
        hv[1] = kok ? (_Float16)f.y : (_Float16)0.0f;
        hv[2] = kok ? (_Float16)f.z : (_Float16)0.0f;
        hv[3] = kok ? (_Float16)f.w : (_Float16)0.0f;
      } else {
        #pragma unroll
        for (int j = 0; j < 4; ++j) {
          bool nok = (gn + j) < N;
          int gnc = nok ? (gn + j) : (N - 1);
          float v = Wt[(long long)gkc * N + gnc];
          hv[j] = (kok && nok) ? (_Float16)v : (_Float16)0.0f;
        }
      }
      #pragma unroll
      for (int j = 0; j < 4; ++j) Bst[c + j][r] = hv[j];
    }
    __syncthreads();

    // ---- fragments (ISA 16-bit wave32 layouts), contiguous 16B LDS runs ----
    const int kbA = (lane < 16) ? 0 : 8;    // A: halves 0-7 -> K=kb..kb+7, 8-15 -> K=kb+16..kb+23
    const int kbB = (lane < 16) ? 0 : 8;    // B: halves 0-7 -> K=kb..kb+7, 8-15 -> K=kb+16..kb+23
    v16h af[2], bf[2];
    #pragma unroll
    for (int im = 0; im < 2; ++im) {
      int row = wm * 32 + im * 16 + l15;
      af[im] = frag_cat(&As[row][kbA], &As[row][kbA + 16]);
    }
    #pragma unroll
    for (int in_ = 0; in_ < 2; ++in_) {
      int col = wn * 32 + in_ * 16 + l15;
      bf[in_] = frag_cat(&Bst[col][kbB], &Bst[col][kbB + 16]);
    }

    #pragma unroll
    for (int im = 0; im < 2; ++im)
      #pragma unroll
      for (int in_ = 0; in_ < 2; ++in_)
        acc[im][in_] = __builtin_amdgcn_wmma_f32_16x16x32_f16(
            false, af[im], false, bf[in_], (short)0, acc[im][in_], false, false);
    __syncthreads();
  }

  // ---- epilogue: C/D layout lanes 0-15 -> M=r, lanes 16-31 -> M=r+8 --------
  #pragma unroll
  for (int im = 0; im < 2; ++im) {
    int mb = m0 + wm * 32 + im * 16 + ((lane >= 16) ? 8 : 0);
    #pragma unroll
    for (int in_ = 0; in_ < 2; ++in_) {
      int n = n0 + wn * 32 + in_ * 16 + l15;
      if (n < N) {
        float bv = bias ? bias[n] : 0.0f;
        #pragma unroll
        for (int r = 0; r < 8; ++r) {
          int m = mb + r;
          if (m < M) {
            float v = acc[im][in_][r] + bv;
            if (act == 1) v = v > 0.0f ? v : 0.0f;
            else if (act == 2) v = tanhf(v);
            C_[(long long)m * N + n] = v;
          }
        }
      }
    }
  }
}

static void gemm_launch(hipStream_t s, const float* A, const float* Wt, const float* bias,
                        float* C, int M, int N, int K, int act) {
  dim3 g((N + GEMM_BN - 1) / GEMM_BN, (M + GEMM_BM - 1) / GEMM_BM);
  gemm_wmma_kernel<<<g, 256, 0, s>>>(A, Wt, bias, C, M, N, K, act);
}

// ------------------------------ GAT kernels --------------------------------

// outn[n*2+h] = sum_c xp[n, h*C + c] * avec[h*C + c]   (one wave32 per (n,h))
__global__ void attn_scores_kernel(const float* __restrict__ xp, const float* __restrict__ avec,
                                   float* __restrict__ outn, int N, int C) {
  long long wid = ((long long)blockIdx.x * blockDim.x + threadIdx.x) >> 5;
  long long wst = ((long long)gridDim.x * blockDim.x) >> 5;
  int lane = threadIdx.x & 31;
  long long total = (long long)N * 2;
  for (; wid < total; wid += wst) {
    int n = (int)(wid >> 1), h = (int)(wid & 1);
    const float* xr = xp + (long long)n * 2 * C + (long long)h * C;
    const float* ar = avec + (long long)h * C;
    float sum = 0.0f;
    for (int c = lane; c < C; c += 32) sum += xr[c] * ar[c];
    #pragma unroll
    for (int off = 16; off > 0; off >>= 1) sum += __shfl_xor(sum, off, 32);
    if (lane == 0) outn[wid] = sum;
  }
}

__global__ void edge_logit_max_kernel(const int* __restrict__ ei, int E, int N,
                                      const float* __restrict__ as_, const float* __restrict__ ad_,
                                      float* __restrict__ logit, float* __restrict__ mseg) {
  long long i = (long long)blockIdx.x * blockDim.x + threadIdx.x;
  long long st = (long long)gridDim.x * blockDim.x;
  long long total = (long long)(E + N) * 2;
  for (; i < total; i += st) {
    int h = (int)(i & 1);
    long long e = i >> 1;
    int sIdx, dIdx;
    if (e < E) { sIdx = ei[e]; dIdx = ei[(long long)E + e]; }
    else       { sIdx = dIdx = (int)(e - E); }
    float l = as_[(long long)sIdx * 2 + h] + ad_[(long long)dIdx * 2 + h];
    l = l > 0.0f ? l : 0.2f * l;                  // leaky_relu(0.2)
    logit[e * 2 + h] = l;
    atomicMaxF(&mseg[(long long)dIdx * 2 + h], l);
  }
}

__global__ void edge_exp_sum_kernel(const int* __restrict__ ei, int E, int N,
                                    float* __restrict__ logit, const float* __restrict__ mseg,
                                    float* __restrict__ sseg) {
  long long i = (long long)blockIdx.x * blockDim.x + threadIdx.x;
  long long st = (long long)gridDim.x * blockDim.x;
  long long total = (long long)(E + N) * 2;
  for (; i < total; i += st) {
    int h = (int)(i & 1);
    long long e = i >> 1;
    int dIdx;
    if (e < E) dIdx = ei[(long long)E + e];
    else       dIdx = (int)(e - E);
    float ev = expf(logit[e * 2 + h] - mseg[(long long)dIdx * 2 + h]);
    logit[e * 2 + h] = ev;
    atomicAdd(&sseg[(long long)dIdx * 2 + h], ev);
  }
}

__global__ void edge_aggr_kernel(const int* __restrict__ ei, int E, int N, int C,
                                 const float* __restrict__ ev, const float* __restrict__ sseg,
                                 const float* __restrict__ xp, float* __restrict__ acc) {
  long long i = (long long)blockIdx.x * blockDim.x + threadIdx.x;
  long long st = (long long)gridDim.x * blockDim.x;
  long long total = (long long)(E + N) * C;
  for (; i < total; i += st) {
    long long e = i / C;
    int c = (int)(i - e * C);
    int sIdx, dIdx;
    if (e < E) { sIdx = ei[e]; dIdx = ei[(long long)E + e]; }
    else       { sIdx = dIdx = (int)(e - E); }
    #pragma unroll
    for (int h = 0; h < 2; ++h) {
      float alpha = ev[e * 2 + h] / (sseg[(long long)dIdx * 2 + h] + 1e-16f);
      float v = xp[((long long)sIdx * 2 + h) * C + c] * alpha;
      atomicAdd(&acc[((long long)dIdx * 2 + h) * C + c], v);
    }
  }
}

__global__ void gat_final_kernel(const float* __restrict__ acc, const float* __restrict__ bvec,
                                 float* __restrict__ out, long long N, int C, int relu_out) {
  long long i = (long long)blockIdx.x * blockDim.x + threadIdx.x;
  long long st = (long long)gridDim.x * blockDim.x;
  long long total = N * C;
  for (; i < total; i += st) {
    long long n = i / C;
    int c = (int)(i - n * C);
    float v = 0.5f * (acc[n * 2 * C + c] + acc[n * 2 * C + C + c]) + bvec[c];
    if (relu_out) v = v > 0.0f ? v : 0.0f;
    out[i] = v;
  }
}

// z = sigmoid(t1 + t2 + bias[c]); cur = z*xnew + (1-z)*cur
__global__ void gate_kernel(const float* __restrict__ xnew, const float* __restrict__ t1,
                            const float* __restrict__ t2, const float* __restrict__ brow,
                            float* __restrict__ cur, long long total, int C) {
  long long i = (long long)blockIdx.x * blockDim.x + threadIdx.x;
  long long st = (long long)gridDim.x * blockDim.x;
  for (; i < total; i += st) {
    int c = (int)(i % C);
    float z = 1.0f / (1.0f + expf(-(t1[i] + t2[i] + brow[c])));
    cur[i] = z * xnew[i] + (1.0f - z) * cur[i];
  }
}

// ------------------------------ GCN kernels --------------------------------

__global__ void gcn_deg_kernel(const int* __restrict__ ei, const float* __restrict__ ew,
                               int E, float* __restrict__ deg) {
  long long i = (long long)blockIdx.x * blockDim.x + threadIdx.x;
  long long st = (long long)gridDim.x * blockDim.x;
  for (; i < E; i += st) atomicAdd(&deg[ei[(long long)E + i]], ew[i]);
}

__global__ void gcn_aggr_kernel(const int* __restrict__ ei, const float* __restrict__ ew, int E,
                                const float* __restrict__ deg, const float* __restrict__ xw,
                                float* __restrict__ acc, int C) {
  long long i = (long long)blockIdx.x * blockDim.x + threadIdx.x;
  long long st = (long long)gridDim.x * blockDim.x;
  long long total = (long long)E * C;
  for (; i < total; i += st) {
    long long e = i / C;
    int c = (int)(i - e * C);
    int sIdx = ei[e], dIdx = ei[(long long)E + e];
    float norm = rsqrtf(deg[sIdx]) * ew[e] * rsqrtf(deg[dIdx]);
    atomicAdd(&acc[(long long)dIdx * C + c], xw[(long long)sIdx * C + c] * norm);
  }
}

// out = relu(acc + xw * dinv^2 (self loop) + b)
__global__ void gcn_final_kernel(const float* __restrict__ acc, const float* __restrict__ xw,
                                 const float* __restrict__ deg, const float* __restrict__ bvec,
                                 float* __restrict__ out, long long total, int C) {
  long long i = (long long)blockIdx.x * blockDim.x + threadIdx.x;
  long long st = (long long)gridDim.x * blockDim.x;
  for (; i < total; i += st) {
    long long n = i / C;
    int c = (int)(i - n * C);
    float dinv2 = 1.0f / deg[n];            // (1/sqrt(d))^2
    float v = acc[i] + xw[i] * dinv2 + bvec[c];
    out[i] = v > 0.0f ? v : 0.0f;
  }
}

// --------------------------- Hypergraph kernels ----------------------------

__global__ void hyper_count_kernel(const int* __restrict__ hei, int E,
                                   float* __restrict__ Dc, float* __restrict__ Bc) {
  long long i = (long long)blockIdx.x * blockDim.x + threadIdx.x;
  long long st = (long long)gridDim.x * blockDim.x;
  for (; i < E; i += st) {
    atomicAdd(&Dc[hei[i]], 1.0f);
    atomicAdd(&Bc[hei[(long long)E + i]], 1.0f);
  }
}

__global__ void hyper_scat1_kernel(const int* __restrict__ hei, int E,
                                   const float* __restrict__ xw, float* __restrict__ eb) {
  long long i = (long long)blockIdx.x * blockDim.x + threadIdx.x;
  long long st = (long long)gridDim.x * blockDim.x;
  long long total = (long long)E * 128;
  for (; i < total; i += st) {
    long long e = i >> 7;
    int c = (int)(i & 127);
    int row = hei[e], col = hei[(long long)E + e];
    atomicAdd(&eb[(long long)col * 128 + c], xw[(long long)row * 128 + c]);
  }
}

__global__ void hyper_scat2_kernel(const int* __restrict__ hei, int E,
                                   const float* __restrict__ eb, const float* __restrict__ Bc,
                                   float* __restrict__ ac2) {
  long long i = (long long)blockIdx.x * blockDim.x + threadIdx.x;
  long long st = (long long)gridDim.x * blockDim.x;
  long long total = (long long)E * 128;
  for (; i < total; i += st) {
    long long e = i >> 7;
    int c = (int)(i & 127);
    int row = hei[e], col = hei[(long long)E + e];
    float bc = Bc[col];
    float binv = bc > 0.0f ? 1.0f / bc : 0.0f;
    atomicAdd(&ac2[(long long)row * 128 + c], eb[(long long)col * 128 + c] * binv);
  }
}

__global__ void hyper_final_kernel(const float* __restrict__ ac2, const float* __restrict__ Dc,
                                   const float* __restrict__ bvec, float* __restrict__ out,
                                   long long total, int C) {
  long long i = (long long)blockIdx.x * blockDim.x + threadIdx.x;
  long long st = (long long)gridDim.x * blockDim.x;
  for (; i < total; i += st) {
    long long n = i / C;
    int c = (int)(i - n * C);
    float d = Dc[n];
    float dinv = d > 0.0f ? 1.0f / d : 0.0f;
    float v = ac2[i] * dinv + bvec[c];
    out[i] = v > 0.0f ? v : 0.0f;
  }
}

// ----------------------------- misc kernels --------------------------------

__global__ void concat2_kernel(const float* __restrict__ a, int Fa, const float* __restrict__ b,
                               int Fb, float* __restrict__ out, long long N) {
  int F = Fa + Fb;
  long long i = (long long)blockIdx.x * blockDim.x + threadIdx.x;
  long long st = (long long)gridDim.x * blockDim.x;
  long long total = N * F;
  for (; i < total; i += st) {
    long long n = i / F;
    int c = (int)(i - n * F);
    out[i] = (c < Fa) ? a[n * Fa + c] : b[n * Fb + (c - Fa)];
  }
}

__global__ void pool_max_kernel(const float* __restrict__ x, const int* __restrict__ batch,
                                long long N, int F, float* __restrict__ out) {
  long long i = (long long)blockIdx.x * blockDim.x + threadIdx.x;
  long long st = (long long)gridDim.x * blockDim.x;
  long long total = N * F;
  for (; i < total; i += st) {
    long long n = i / F;
    int c = (int)(i - n * F);
    atomicMaxF(&out[(long long)batch[n] * F + c], x[i]);
  }
}

__global__ void att_softmax_kernel(const float* __restrict__ sx, const float* __restrict__ st_,
                                   float* __restrict__ out_att) {
  int b = blockIdx.x * blockDim.x + threadIdx.x;
  if (b < NB) {
    float a0 = sx[b], a1 = st_[b];
    float mx = fmaxf(a0, a1);
    float e0 = expf(a0 - mx), e1 = expf(a1 - mx);
    float s = e0 + e1;
    out_att[b * 2 + 0] = e0 / s;
    out_att[b * 2 + 1] = e1 / s;
  }
}

__global__ void xc_build_kernel(const float* __restrict__ att, const float* __restrict__ embm,
                                const float* __restrict__ embt, float* __restrict__ xc) {
  long long i = (long long)blockIdx.x * blockDim.x + threadIdx.x;
  long long total = (long long)NB * 256;
  if (i < total) {
    long long b = i >> 8;
    int c = (int)(i & 255);
    xc[i] = (c < 128) ? att[b * 2 + 0] * embm[b * 128 + c]
                      : att[b * 2 + 1] * embt[b * 128 + (c - 128)];
  }
}

// ------------------------------- host helpers ------------------------------

static void run_gat(hipStream_t s, const float* x, int N, int Fin, int C,
                    const int* ei, int E,
                    const float* w, const float* as_, const float* ad_, const float* bv,
                    float* out, int relu_out,
                    float* XP, float* ACC, float* LOG, float* ASR, float* ADS,
                    float* MSG, float* SSG)
{
  gemm_launch(s, x, w, nullptr, XP, N, 2 * C, Fin, 0);             // xp = x@w
  attn_scores_kernel<<<nblk((long long)N * 64), 256, 0, s>>>(XP, as_, ASR, N, C);
  attn_scores_kernel<<<nblk((long long)N * 64), 256, 0, s>>>(XP, ad_, ADS, N, C);
  fill_kernel<<<nblk((long long)N * 2), 256, 0, s>>>(MSG, -1e30f, (long long)N * 2);
  fill_kernel<<<nblk((long long)N * 2), 256, 0, s>>>(SSG, 0.0f, (long long)N * 2);
  edge_logit_max_kernel<<<nblk((long long)(E + N) * 2), 256, 0, s>>>(ei, E, N, ASR, ADS, LOG, MSG);
  edge_exp_sum_kernel<<<nblk((long long)(E + N) * 2), 256, 0, s>>>(ei, E, N, LOG, MSG, SSG);
  fill_kernel<<<nblk((long long)N * 2 * C), 256, 0, s>>>(ACC, 0.0f, (long long)N * 2 * C);
  edge_aggr_kernel<<<nblk((long long)(E + N) * C), 256, 0, s>>>(ei, E, N, C, LOG, SSG, XP, ACC);
  gat_final_kernel<<<nblk((long long)N * C), 256, 0, s>>>(ACC, bv, out, (long long)N, C, relu_out);
}

static void run_hyper(hipStream_t s, const float* hxin, int Fin, const int* hei, int E, int N,
                      const float* w, const float* bv, float* out,
                      float* XW, float* EB, float* AC2, float* Dc, float* Bc)
{
  gemm_launch(s, hxin, w, nullptr, XW, N, 128, Fin, 0);
  fill_kernel<<<nblk(N), 256, 0, s>>>(Dc, 0.0f, N);
  fill_kernel<<<nblk(N), 256, 0, s>>>(Bc, 0.0f, N);
  hyper_count_kernel<<<nblk(E), 256, 0, s>>>(hei, E, Dc, Bc);
  fill_kernel<<<nblk((long long)N * 128), 256, 0, s>>>(EB, 0.0f, (long long)N * 128);
  hyper_scat1_kernel<<<nblk((long long)E * 128), 256, 0, s>>>(hei, E, XW, EB);
  fill_kernel<<<nblk((long long)N * 128), 256, 0, s>>>(AC2, 0.0f, (long long)N * 128);
  hyper_scat2_kernel<<<nblk((long long)E * 128), 256, 0, s>>>(hei, E, EB, Bc, AC2);
  hyper_final_kernel<<<nblk((long long)N * 128), 256, 0, s>>>(AC2, Dc, bv, out, (long long)N * 128, 128);
}

// --------------------------------- driver ----------------------------------

extern "C" void kernel_launch(void* const* d_in, const int* in_sizes, int n_in,
                              void* d_out, int out_size, void* d_ws, size_t ws_size,
                              hipStream_t stream)
{
  (void)in_sizes; (void)n_in; (void)out_size; (void)ws_size;

  // params in shapes-dict order (indices 0..55), then the 8 data inputs
  enum {
    GCN_W = 0, GCN_B, PG1_W, PG1_AS, PG1_AD, PG1_B, PG2_W, PG2_AS, PG2_AD, PG2_B,
    PFC1_W, PFC1_B, PFC2_W, PFC2_B, P_BIAS,
    MG0_W, MG0_AS, MG0_AD, MG0_B, MG1_W, MG1_AS, MG1_AD, MG1_B,
    MG2_W, MG2_AS, MG2_AD, MG2_B,
    MFC1_W, MFC1_B, MFC2_W, MFC2_B, M_BIAS,
    HY1_W, HY1_B, HY2_W, HY2_B,
    G1M_W, G1M_B, G2M_W, G2M_B, G1P_W, G1P_B, G2P_W, G2P_B,
    AX_W1, AX_B1, AX_W2, AT_W1, AT_B1, AT_W2,
    FC1_W, FC1_B, FC2_W, FC2_B, OUT_W, OUT_B
  };
  const float* P[56];
  for (int i = 0; i < 56; ++i) P[i] = (const float*)d_in[i];
  const float* mol_x     = (const float*)d_in[56];
  const int*   mol_ei    = (const int*)  d_in[57];
  /* d_in[58] = hyper_edge: accepted but unused (reference uses mol_edge_index) */
  const int*   mol_batch = (const int*)  d_in[59];
  const float* tgt_x     = (const float*)d_in[60];
  const int*   tgt_ei    = (const int*)  d_in[61];
  const float* tgt_w     = (const float*)d_in[62];
  const int*   tgt_batch = (const int*)  d_in[63];
  float* dout = (float*)d_out;   // [0..127] out, [128..383] att

  // ---- workspace arena (floats) ----
  float* ws = (float*)d_ws;
  size_t o = 0;
  auto alloc = [&](size_t n) { float* p = ws + o; o += n; return p; };
  float* XP    = alloc(15360000);   // gemm xp / gcn xw  (max Nm*512, Np*256)
  float* ACC   = alloc(15360000);   // gat head-acc / gcn acc / gate t1,t2 / hyper scratch
  float* OUT   = alloc(7680000);    // gat layer output / hx2
  float* MOL   = alloc(7680000);    // molecule running features [Nm,256]
  float* TB    = alloc(7680000);    // protein running features  [Np,128]
  float* LOG   = alloc(720000);     // edge logits/exp [(E+N)*2]
  float* ASR   = alloc(120000);
  float* ADS   = alloc(120000);
  float* MSG   = alloc(120000);
  float* SSG   = alloc(120000);
  float* MFEAT = alloc(11520000);   // mol_feat [Nm,384]
  float* DEG   = alloc(60000);      // gcn deg / hyper D
  float* BC    = alloc(30000);      // hyper B
  float* MG    = alloc((size_t)NB * 384);
  float* TG    = alloc((size_t)NB * 128);
  float* H1    = alloc((size_t)NB * 1024);
  float* EMBM  = alloc((size_t)NB * 128);
  float* EMBT  = alloc((size_t)NB * 128);
  float* TX    = alloc((size_t)NB * 64);
  float* SX    = alloc(NB);
  float* ST    = alloc(NB);
  float* XC    = alloc((size_t)NB * 256);
  float* HH2   = alloc((size_t)NB * 1024);
  float* HH3   = alloc((size_t)NB * 512);

  // aliases for hyper phase (XP/ACC/OUT free between mol GATs and protein)
  float* HX  = XP;                // [Nm,334]
  float* XWH = ACC;               // [Nm,128]
  float* EB  = ACC + 3840000;     // [Nm,128]
  float* AC2 = ACC + 7680000;     // [Nm,128]
  float* HX1 = ACC + 11520000;    // [Nm,128]
  float* HX2 = OUT;               // [Nm,128]
  float* T1  = ACC;               // gate temps (after gat_final, ACC is free)
  float* T2  = ACC + 7680000;

  hipStream_t s = stream;

  // ================= molecule encoder: 3 GAT + gated residual ==============
  // mol = relu(gat0(mol_x))
  run_gat(s, mol_x, NM, 78, 256, mol_ei, EM,
          P[MG0_W], P[MG0_AS], P[MG0_AD], P[MG0_B], MOL, 1,
          XP, ACC, LOG, ASR, ADS, MSG, SSG);
  // i = 1 (relu on x), i = 2 (no relu)
  for (int i = 1; i <= 2; ++i) {
    const float* w  = (i == 1) ? P[MG1_W]  : P[MG2_W];
    const float* as_= (i == 1) ? P[MG1_AS] : P[MG2_AS];
    const float* ad_= (i == 1) ? P[MG1_AD] : P[MG2_AD];
    const float* bv = (i == 1) ? P[MG1_B]  : P[MG2_B];
    run_gat(s, MOL, NM, 256, 256, mol_ei, EM, w, as_, ad_, bv, OUT, (i < 2) ? 1 : 0,
            XP, ACC, LOG, ASR, ADS, MSG, SSG);
    gemm_launch(s, OUT, P[MFC1_W], P[MFC1_B], T1, NM, 256, 256, 0);
    gemm_launch(s, MOL, P[MFC2_W], P[MFC2_B], T2, NM, 256, 256, 0);
    gate_kernel<<<nblk((long long)NM * 256), 256, 0, s>>>(OUT, T1, T2, P[M_BIAS],
                                                          MOL, (long long)NM * 256, 256);
  }

  // ====================== hypergraph branch (2 convs) =======================
  concat2_kernel<<<nblk((long long)NM * 334), 256, 0, s>>>(MOL, 256, mol_x, 78, HX, NM);
  run_hyper(s, HX,  334, mol_ei, EM, NM, P[HY1_W], P[HY1_B], HX1, XWH, EB, AC2, DEG, BC);
  run_hyper(s, HX1, 128, mol_ei, EM, NM, P[HY2_W], P[HY2_B], HX2, XWH, EB, AC2, DEG, BC);
  concat2_kernel<<<nblk((long long)NM * 384), 256, 0, s>>>(MOL, 256, HX2, 128, MFEAT, NM);

  // ================= protein encoder: GCN + 2 GAT + gating ==================
  fill_kernel<<<nblk(NP), 256, 0, s>>>(DEG, 1.0f, NP);           // self-loop weight 1
  gcn_deg_kernel<<<nblk(EP), 256, 0, s>>>(tgt_ei, tgt_w, EP, DEG);
  gemm_launch(s, tgt_x, P[GCN_W], nullptr, XP, NP, 128, 33, 0);  // xw
  fill_kernel<<<nblk((long long)NP * 128), 256, 0, s>>>(ACC, 0.0f, (long long)NP * 128);
  gcn_aggr_kernel<<<nblk((long long)EP * 128), 256, 0, s>>>(tgt_ei, tgt_w, EP, DEG, XP, ACC, 128);
  gcn_final_kernel<<<nblk((long long)NP * 128), 256, 0, s>>>(ACC, XP, DEG, P[GCN_B], TB,
                                                             (long long)NP * 128, 128);
  for (int i = 1; i <= 2; ++i) {
    const float* w  = (i == 1) ? P[PG1_W]  : P[PG2_W];
    const float* as_= (i == 1) ? P[PG1_AS] : P[PG2_AS];
    const float* ad_= (i == 1) ? P[PG1_AD] : P[PG2_AD];
    const float* bv = (i == 1) ? P[PG1_B]  : P[PG2_B];
    run_gat(s, TB, NP, 128, 128, tgt_ei, EP, w, as_, ad_, bv, OUT, (i < 2) ? 1 : 0,
            XP, ACC, LOG, ASR, ADS, MSG, SSG);
    gemm_launch(s, OUT, P[PFC1_W], P[PFC1_B], T1, NP, 128, 128, 0);
    gemm_launch(s, TB,  P[PFC2_W], P[PFC2_B], T2, NP, 128, 128, 0);
    gate_kernel<<<nblk((long long)NP * 128), 256, 0, s>>>(OUT, T1, T2, P[P_BIAS],
                                                          TB, (long long)NP * 128, 128);
  }

  // ======================= global max pool + MLP head =======================
  fill_kernel<<<nblk((long long)NB * 384), 256, 0, s>>>(MG, -1e30f, (long long)NB * 384);
  pool_max_kernel<<<nblk((long long)NM * 384), 256, 0, s>>>(MFEAT, mol_batch, NM, 384, MG);
  fill_kernel<<<nblk((long long)NB * 128), 256, 0, s>>>(TG, -1e30f, (long long)NB * 128);
  pool_max_kernel<<<nblk((long long)NP * 128), 256, 0, s>>>(TB, tgt_batch, NP, 128, TG);

  gemm_launch(s, MG, P[G1M_W], P[G1M_B], H1, NB, 1024, 384, 1);
  gemm_launch(s, H1, P[G2M_W], P[G2M_B], EMBM, NB, 128, 1024, 0);
  gemm_launch(s, TG, P[G1P_W], P[G1P_B], H1, NB, 1024, 128, 1);
  gemm_launch(s, H1, P[G2P_W], P[G2P_B], EMBT, NB, 128, 1024, 0);

  gemm_launch(s, EMBM, P[AX_W1], P[AX_B1], TX, NB, 64, 128, 2);  // tanh
  gemm_launch(s, TX,   P[AX_W2], nullptr,  SX, NB, 1, 64, 0);
  gemm_launch(s, EMBT, P[AT_W1], P[AT_B1], TX, NB, 64, 128, 2);  // tanh
  gemm_launch(s, TX,   P[AT_W2], nullptr,  ST, NB, 1, 64, 0);

  att_softmax_kernel<<<1, NB, 0, s>>>(SX, ST, dout + NB);        // att -> d_out[128..383]
  xc_build_kernel<<<nblk((long long)NB * 256), 256, 0, s>>>(dout + NB, EMBM, EMBT, XC);

  gemm_launch(s, XC,  P[FC1_W], P[FC1_B], HH2, NB, 1024, 256, 1);
  gemm_launch(s, HH2, P[FC2_W], P[FC2_B], HH3, NB, 512, 1024, 1);
  gemm_launch(s, HH3, P[OUT_W], P[OUT_B], dout, NB, 1, 512, 0);  // out -> d_out[0..127]
}
